// AttentionBlock_26963804684476
// MI455X (gfx1250) — compile-verified, matrix operand order checked by reading, not run on hardware
//
#include <hip/hip_runtime.h>

// ---------------------------------------------------------------------------
// Types for CDNA5 WMMA bf16: D(f32 16x16) = A(bf16 16x32) * B(bf16 32x16) + C
// ---------------------------------------------------------------------------
typedef __attribute__((ext_vector_type(16))) __bf16 v16bf;
typedef __attribute__((ext_vector_type(8)))  float  v8f;
typedef __attribute__((ext_vector_type(4)))  unsigned int v4u;
typedef __attribute__((ext_vector_type(4)))  int    v4i;
typedef __attribute__((ext_vector_type(8)))  int    v8i;

#if __has_builtin(__builtin_amdgcn_tensor_load_to_lds) && \
    __has_builtin(__builtin_amdgcn_s_wait_tensorcnt)
#define HAVE_TDM 1
#else
#define HAVE_TDM 0
#endif

union FragBF {
  unsigned int u[8];   // 8 dwords = 16 bf16 per lane
  v16bf v;
};

__device__ __forceinline__ unsigned short f2bf(float f) {
  unsigned int u = __float_as_uint(f);
  u += 0x7FFFu + ((u >> 16) & 1u);   // round-to-nearest-even
  return (unsigned short)(u >> 16);
}

// Load a 16x32 bf16 WMMA A/B fragment from a (row, K) row-major bf16 matrix.
// Lane mapping per CDNA5 ISA 7.12.2 (16-bit A 16x32):
//   lane L<16 : row = row0+L, K = {k0+0..7} (vgpr0-3) and {k0+16..23} (vgpr4-7)
//   lane L+16 : same row,     K = {k0+8..15} and {k0+24..31}
// U = matrix base viewed as uint (2 bf16 / uint); ld_u = row stride in uints;
// k0 must be even.
__device__ __forceinline__ FragBF load_frag(const unsigned int* __restrict__ U,
                                            int row0, int ld_u, int k0, int lane) {
  FragBF f;
  const int r    = lane & 15;
  const int half = lane >> 4;
  const unsigned int* p = U + (size_t)(row0 + r) * (size_t)ld_u + ((k0 + half * 8) >> 1);
#pragma unroll
  for (int i = 0; i < 4; ++i) {
    f.u[i]     = p[i];       // K = kbase + 2i, 2i+1
    f.u[4 + i] = p[8 + i];   // K = kbase + 16 + 2i, ...
  }
  return f;
}

__device__ __forceinline__ v8f wmma_bf16(const FragBF& a, const FragBF& b, v8f c) {
  return __builtin_amdgcn_wmma_f32_16x16x32_bf16(false, a.v, false, b.v,
                                                 (short)0, c, false, false);
}

#if HAVE_TDM
// ---------------------------------------------------------------------------
// Tensor Data Mover: 1D/2D bf16 tile load Global -> LDS.
// D# packing per CDNA5 ISA 8.3/8.4:
//  group0: [1:0]=count=1, [63:32]=lds_addr, [120:64]=global_addr, [127:126]=2
//  group1: [17:16]=data_size(1 => 2B), [79:48]=tensor_dim0, [111:80]=tensor_dim1,
//          [127:112]=tile_dim0, [143:128]=tile_dim1, [207:160]=tensor_dim0_stride
// tile_dim1 == 0 -> 1D transfer.
// ---------------------------------------------------------------------------
__device__ __forceinline__ void tdm_load_2d(unsigned lds_addr, unsigned long long gaddr,
                                            unsigned tensor_d0, unsigned tensor_d1,
                                            unsigned tile_d0, unsigned tile_d1,
                                            unsigned long long stride0) {
  v4u g0;
  g0[0] = 1u;                                                // count=1, user mode
  g0[1] = lds_addr;                                          // LDS byte address
  g0[2] = (unsigned)(gaddr & 0xFFFFFFFFull);                 // global_addr[31:0]
  g0[3] = (unsigned)((gaddr >> 32) & 0x1FFFFFFull)           // global_addr[56:32]
        | (2u << 30);                                        // type = 2 ("image")
  unsigned w0 = 1u << 16;                                    // data_size = 2 bytes
  unsigned w1 = (tensor_d0 & 0xFFFFu) << 16;
  unsigned w2 = (tensor_d0 >> 16) | ((tensor_d1 & 0xFFFFu) << 16);
  unsigned w3 = (tensor_d1 >> 16) | ((tile_d0 & 0xFFFFu) << 16);
  unsigned w4 = (tile_d1 & 0xFFFFu);                         // tile_dim2 = 0
  unsigned w5 = (unsigned)(stride0 & 0xFFFFFFFFull);
  unsigned w6 = (unsigned)((stride0 >> 32) & 0xFFFFull);     // dim1_stride lo = 0
  unsigned w7 = 0u;
  v8i g1;
  g1[0] = (int)w0; g1[1] = (int)w1; g1[2] = (int)w2; g1[3] = (int)w3;
  g1[4] = (int)w4; g1[5] = (int)w5; g1[6] = (int)w6; g1[7] = (int)w7;
  v4i z4 = {0, 0, 0, 0};
#if __clang_major__ >= 23
  v8i z8 = {0, 0, 0, 0, 0, 0, 0, 0};
  __builtin_amdgcn_tensor_load_to_lds(g0, g1, z4, z4, z8, 0);
#else
  __builtin_amdgcn_tensor_load_to_lds(g0, g1, z4, z4, 0);
#endif
}
#endif  // HAVE_TDM

// ---------------------------------------------------------------------------
// Problem constants
// ---------------------------------------------------------------------------
#define BATCH   16
#define CH      512
#define HW      1024
#define HEADS   8
#define HD      64
#define GROUPS  8
#define CPG     64          // channels per group
#define GN_N    (CPG * HW)  // 65536 elements per (b, group)

// ---------------------------------------------------------------------------
// Kernel: fp32 -> bf16 weight conversion
// ---------------------------------------------------------------------------
__global__ __launch_bounds__(256) void cvt_bf16_kernel(const float* __restrict__ src,
                                                       unsigned short* __restrict__ dst,
                                                       int n) {
  int i = blockIdx.x * 256 + threadIdx.x;
  if (i < n) dst[i] = f2bf(src[i]);
}

// ---------------------------------------------------------------------------
// Kernel: GroupNorm -> xn stored transposed (b, p, c) in bf16 (N-major for GEMM B)
// ---------------------------------------------------------------------------
__global__ __launch_bounds__(256) void groupnorm_kernel(const float* __restrict__ x,
                                                        const float* __restrict__ gamma,
                                                        const float* __restrict__ beta,
                                                        unsigned short* __restrict__ xn_t) {
  __shared__ float rs[256];
  __shared__ float rq[256];
  const int bg = blockIdx.x;
  const int b  = bg >> 3;
  const int g  = bg & 7;
  const float* __restrict__ xb = x + (size_t)(b * CH + g * CPG) * HW;

  float sum = 0.f, sq = 0.f;
  for (int idx = threadIdx.x; idx < GN_N; idx += 256) {
    float v = xb[idx];
    sum += v;
    sq  += v * v;
  }
  rs[threadIdx.x] = sum;
  rq[threadIdx.x] = sq;
  __syncthreads();
#pragma unroll
  for (int s = 128; s > 0; s >>= 1) {
    if ((int)threadIdx.x < s) {
      rs[threadIdx.x] += rs[threadIdx.x + s];
      rq[threadIdx.x] += rq[threadIdx.x + s];
    }
    __syncthreads();
  }
  const float inv  = 1.0f / (float)GN_N;
  const float mean = rs[0] * inv;
  const float var  = rq[0] * inv - mean * mean;
  const float rstd = rsqrtf(var + 1e-5f);

  for (int idx = threadIdx.x; idx < GN_N; idx += 256) {
    const int cl = idx >> 10;
    const int p  = idx & 1023;
    const int c  = g * CPG + cl;
    const float v = (xb[idx] - mean) * rstd * gamma[c] + beta[c];
    xn_t[(size_t)(b * HW + p) * CH + c] = f2bf(v);
  }
}

// ---------------------------------------------------------------------------
// Kernel: QKV GEMM  qkv[b] = Wqkv(1536x512) @ xn[b](512x1024) + b_qkv
// grid (12, 8, 16), block 128 (4 waves, 2x2 of 64x64 tiles)
// ---------------------------------------------------------------------------
__global__ __launch_bounds__(128) void qkv_gemm_kernel(const unsigned int* __restrict__ wq_u,
                                                       const unsigned int* __restrict__ xn_u,
                                                       const float* __restrict__ b_qkv,
                                                       unsigned short* __restrict__ qb,
                                                       unsigned short* __restrict__ kb,
                                                       unsigned short* __restrict__ vb,
                                                       float scale) {
  const int b    = blockIdx.z;
  const int wave = threadIdx.x >> 5;
  const int lane = threadIdx.x & 31;
  const int m0   = blockIdx.x * 128 + (wave >> 1) * 64;
  const int n0   = blockIdx.y * 128 + (wave & 1) * 64;
  const unsigned int* __restrict__ Bm = xn_u + (size_t)b * (HW * (CH / 2));

  const v8f vz = {0.f, 0.f, 0.f, 0.f, 0.f, 0.f, 0.f, 0.f};
  v8f acc[4][4];
#pragma unroll
  for (int mi = 0; mi < 4; ++mi)
#pragma unroll
    for (int ni = 0; ni < 4; ++ni) acc[mi][ni] = vz;

  for (int k = 0; k < CH; k += 32) {
    FragBF a[4], bf[4];
#pragma unroll
    for (int mi = 0; mi < 4; ++mi) a[mi]  = load_frag(wq_u, m0 + mi * 16, CH / 2, k, lane);
#pragma unroll
    for (int ni = 0; ni < 4; ++ni) bf[ni] = load_frag(Bm,   n0 + ni * 16, CH / 2, k, lane);
#pragma unroll
    for (int mi = 0; mi < 4; ++mi)
#pragma unroll
      for (int ni = 0; ni < 4; ++ni)
        acc[mi][ni] = wmma_bf16(a[mi], bf[ni], acc[mi][ni]);
  }

  const int r    = lane & 15;
  const int half = lane >> 4;
#pragma unroll
  for (int mi = 0; mi < 4; ++mi) {
#pragma unroll
    for (int ni = 0; ni < 4; ++ni) {
#pragma unroll
      for (int e = 0; e < 8; ++e) {
        const int o = m0 + mi * 16 + e + half * 8;
        const int p = n0 + ni * 16 + r;
        float val = acc[mi][ni][e] + b_qkv[o];
        const int sec = o >> 9;
        const int oc  = o & 511;
        const int h   = oc >> 6;
        const int d   = oc & 63;
        const size_t bh = (size_t)(b * HEADS + h);
        if (sec == 0)      qb[(bh * HW + p) * HD + d] = f2bf(val * scale);
        else if (sec == 1) kb[(bh * HW + p) * HD + d] = f2bf(val);
        else               vb[(bh * HD + d) * HW + p] = f2bf(val);
      }
    }
  }
}

// ---------------------------------------------------------------------------
// Kernel: flash attention per (b, h).
// q,k layout (bh, p, d); v layout (bh, d, p); q pre-scaled by hd^-0.5.
// Workgroup = 64 query rows (4 waves x 16 rows); j loop in tiles of 32.
// K-tile (32x64 bf16, contiguous 4KB) and V-tile (64 rows x 32 cols, strided)
// are DMA'd to LDS once per block per j-step via the Tensor Data Mover
// (TENSORcnt path); waves build WMMA fragments from LDS. Falls back to direct
// global fragment loads when the TDM builtin is unavailable.
// grid (16, 128), block 128
// ---------------------------------------------------------------------------
__global__ __launch_bounds__(128) void attn_kernel(const unsigned int* __restrict__ qbuf,
                                                   const unsigned int* __restrict__ kbuf,
                                                   const unsigned int* __restrict__ vbuf,
                                                   unsigned short* __restrict__ attn_out) {
#if HAVE_TDM
  __shared__ unsigned short ldsK[32 * 64];      // rows j (32) x d (64)
  __shared__ unsigned short ldsV[64 * 32];      // rows d (64) x j (32)
#endif
  __shared__ unsigned short ldsP[4][16 * 32];   // per-wave 16x32 bf16 P staging

  const int bh   = blockIdx.y;                  // b*HEADS + h
  const int wave = threadIdx.x >> 5;
  const int lane = threadIdx.x & 31;
  const int r    = lane & 15;
  const int half = lane >> 4;
  const int i0   = blockIdx.x * 64 + wave * 16;

  const unsigned int* __restrict__ q = qbuf + (size_t)bh * (HW * (HD / 2));
  const unsigned int* __restrict__ k = kbuf + (size_t)bh * (HW * (HD / 2));
  const unsigned int* __restrict__ v = vbuf + (size_t)bh * (HD * (HW / 2));

  // Q fragments for K(=d) in [0,32) and [32,64): loaded once, wave-private.
  const FragBF qa0 = load_frag(q, i0, HD / 2, 0,  lane);
  const FragBF qa1 = load_frag(q, i0, HD / 2, 32, lane);

  const v8f vz = {0.f, 0.f, 0.f, 0.f, 0.f, 0.f, 0.f, 0.f};
  v8f o[4];
#pragma unroll
  for (int n = 0; n < 4; ++n) o[n] = vz;
  float m[8], l[8];
#pragma unroll
  for (int e = 0; e < 8; ++e) { m[e] = -1e30f; l[e] = 0.f; }

  unsigned short* __restrict__ lp = &ldsP[wave][0];
  const unsigned int* __restrict__ lpu = (const unsigned int*)lp;

#if HAVE_TDM
  const unsigned long long kg = (unsigned long long)(size_t)k;
  const unsigned long long vg = (unsigned long long)(size_t)v;
  const unsigned ldsK_addr = (unsigned)(size_t)&ldsK[0];
  const unsigned ldsV_addr = (unsigned)(size_t)&ldsV[0];
  const unsigned int* __restrict__ lk = (const unsigned int*)&ldsK[0];
  const unsigned int* __restrict__ lv = (const unsigned int*)&ldsV[0];
#endif

  for (int jt = 0; jt < HW; jt += 32) {
#if HAVE_TDM
    if (wave == 0) {
      // K rows jt..jt+31 are contiguous: 1-D transfer of 2048 bf16 (4KB).
      tdm_load_2d(ldsK_addr, kg + (unsigned long long)jt * (HD * 2),
                  /*tensor_d0=*/32 * HD, /*tensor_d1=*/0,
                  /*tile_d0=*/32 * HD, /*tile_d1=*/0, /*stride0=*/32 * HD);
      // V tile: 64 rows (d) x 32 cols (p), row stride HW elements.
      tdm_load_2d(ldsV_addr, vg + (unsigned long long)jt * 2,
                  /*tensor_d0=*/HW, /*tensor_d1=*/HD,
                  /*tile_d0=*/32, /*tile_d1=*/HD, /*stride0=*/HW);
      __builtin_amdgcn_s_wait_tensorcnt(0);
    }
    __syncthreads();
    const FragBF kb00 = load_frag(lk, 0,  HD / 2, 0,  lane);
    const FragBF kb01 = load_frag(lk, 0,  HD / 2, 32, lane);
    const FragBF kb10 = load_frag(lk, 16, HD / 2, 0,  lane);
    const FragBF kb11 = load_frag(lk, 16, HD / 2, 32, lane);
#else
    const FragBF kb00 = load_frag(k, jt,      HD / 2, 0,  lane);
    const FragBF kb01 = load_frag(k, jt,      HD / 2, 32, lane);
    const FragBF kb10 = load_frag(k, jt + 16, HD / 2, 0,  lane);
    const FragBF kb11 = load_frag(k, jt + 16, HD / 2, 32, lane);
#endif
    v8f s0 = vz, s1 = vz;
    s0 = wmma_bf16(qa0, kb00, s0);
    s0 = wmma_bf16(qa1, kb01, s0);
    s1 = wmma_bf16(qa0, kb10, s1);
    s1 = wmma_bf16(qa1, kb11, s1);

    // Online softmax across these 32 columns. Row (e + half*8) lives in the
    // 16 lanes of each half -> shfl_xor offsets 1..8 stay within the half.
    float alpha[8];
#pragma unroll
    for (int e = 0; e < 8; ++e) {
      float rv = fmaxf(s0[e], s1[e]);
#pragma unroll
      for (int off = 8; off > 0; off >>= 1) rv = fmaxf(rv, __shfl_xor(rv, off, 32));
      const float mn = fmaxf(m[e], rv);
      alpha[e] = __expf(m[e] - mn);
      m[e] = mn;
      const float p0 = __expf(s0[e] - mn);
      const float p1 = __expf(s1[e] - mn);
      s0[e] = p0;
      s1[e] = p1;
      float rsum = p0 + p1;
#pragma unroll
      for (int off = 8; off > 0; off >>= 1) rsum += __shfl_xor(rsum, off, 32);
      l[e] = l[e] * alpha[e] + rsum;
    }

    // C-layout (row = e + half*8, col = lane&15) -> LDS -> A-fragment layout.
#pragma unroll
    for (int e = 0; e < 8; ++e) {
      const int row = e + half * 8;
      lp[row * 32 + r]      = f2bf(s0[e]);
      lp[row * 32 + 16 + r] = f2bf(s1[e]);
    }
    const FragBF pa = load_frag(lpu, 0, 16, 0, lane);   // 16 rows x 32 j's

    // O[n] += P @ V^T[:, n*16 .. n*16+15]
#pragma unroll
    for (int n = 0; n < 4; ++n) {
#if HAVE_TDM
      const FragBF vbf = load_frag(lv, n * 16, 16, 0, lane);
#else
      const FragBF vbf = load_frag(v, n * 16, HW / 2, jt, lane);
#endif
      v8f t = o[n];
#pragma unroll
      for (int e = 0; e < 8; ++e) t[e] *= alpha[e];
      o[n] = wmma_bf16(pa, vbf, t);
    }
#if HAVE_TDM
    __syncthreads();   // all waves done with ldsK/ldsV before next DMA
#endif
  }

  // Normalize by l and store to (b, p, c) bf16.
#pragma unroll
  for (int e = 0; e < 8; ++e) l[e] = 1.0f / l[e];
  const int b = bh >> 3;
  const int h = bh & 7;
#pragma unroll
  for (int n = 0; n < 4; ++n) {
#pragma unroll
    for (int e = 0; e < 8; ++e) {
      const int i = i0 + e + half * 8;
      const int d = n * 16 + r;
      attn_out[(size_t)(b * HW + i) * CH + h * HD + d] = f2bf(o[n][e] * l[e]);
    }
  }
}

// ---------------------------------------------------------------------------
// Kernel: out projection  y = W_out(512x512) @ attn(512x1024) + b_out + x
// grid (4, 8, 16), block 128
// ---------------------------------------------------------------------------
__global__ __launch_bounds__(128) void out_gemm_kernel(const unsigned int* __restrict__ wo_u,
                                                       const unsigned int* __restrict__ ao_u,
                                                       const float* __restrict__ b_out,
                                                       const float* __restrict__ x,
                                                       float* __restrict__ out) {
  const int b    = blockIdx.z;
  const int wave = threadIdx.x >> 5;
  const int lane = threadIdx.x & 31;
  const int m0   = blockIdx.x * 128 + (wave >> 1) * 64;
  const int n0   = blockIdx.y * 128 + (wave & 1) * 64;
  const unsigned int* __restrict__ Bm = ao_u + (size_t)b * (HW * (CH / 2));

  const v8f vz = {0.f, 0.f, 0.f, 0.f, 0.f, 0.f, 0.f, 0.f};
  v8f acc[4][4];
#pragma unroll
  for (int mi = 0; mi < 4; ++mi)
#pragma unroll
    for (int ni = 0; ni < 4; ++ni) acc[mi][ni] = vz;

  for (int k = 0; k < CH; k += 32) {
    FragBF a[4], bf[4];
#pragma unroll
    for (int mi = 0; mi < 4; ++mi) a[mi]  = load_frag(wo_u, m0 + mi * 16, CH / 2, k, lane);
#pragma unroll
    for (int ni = 0; ni < 4; ++ni) bf[ni] = load_frag(Bm,   n0 + ni * 16, CH / 2, k, lane);
#pragma unroll
    for (int mi = 0; mi < 4; ++mi)
#pragma unroll
      for (int ni = 0; ni < 4; ++ni)
        acc[mi][ni] = wmma_bf16(a[mi], bf[ni], acc[mi][ni]);
  }

  const int r    = lane & 15;
  const int half = lane >> 4;
#pragma unroll
  for (int mi = 0; mi < 4; ++mi) {
#pragma unroll
    for (int ni = 0; ni < 4; ++ni) {
#pragma unroll
      for (int e = 0; e < 8; ++e) {
        const int o = m0 + mi * 16 + e + half * 8;
        const int p = n0 + ni * 16 + r;
        const size_t idx = (size_t)(b * CH + o) * HW + p;
        out[idx] = acc[mi][ni][e] + b_out[o] + x[idx];
      }
    }
  }
}

// ---------------------------------------------------------------------------
// Host launcher
// ---------------------------------------------------------------------------
extern "C" void kernel_launch(void* const* d_in, const int* in_sizes, int n_in,
                              void* d_out, int out_size, void* d_ws, size_t ws_size,
                              hipStream_t stream) {
  (void)in_sizes; (void)n_in; (void)out_size; (void)ws_size;

  const float* x      = (const float*)d_in[0];
  const float* gamma  = (const float*)d_in[1];
  const float* beta   = (const float*)d_in[2];
  const float* w_qkv  = (const float*)d_in[3];
  const float* b_qkv  = (const float*)d_in[4];
  const float* w_out  = (const float*)d_in[5];
  const float* b_out  = (const float*)d_in[6];
  float* out = (float*)d_out;

  // Workspace carve-up (bf16 buffers stored as ushort). Total ~86 MB.
  char* ws = (char*)d_ws;
  unsigned short* wq_bf = (unsigned short*)ws;  ws += (size_t)3 * CH * CH * 2;
  unsigned short* wo_bf = (unsigned short*)ws;  ws += (size_t)CH * CH * 2;
  unsigned short* xn_t  = (unsigned short*)ws;  ws += (size_t)BATCH * HW * CH * 2;
  unsigned short* qb    = (unsigned short*)ws;  ws += (size_t)BATCH * CH * HW * 2;
  unsigned short* kb    = (unsigned short*)ws;  ws += (size_t)BATCH * CH * HW * 2;
  unsigned short* vb    = (unsigned short*)ws;  ws += (size_t)BATCH * CH * HW * 2;
  unsigned short* ao    = (unsigned short*)ws;

  const float scale = 0.125f;  // HD^-0.5 = 64^-0.5

  cvt_bf16_kernel<<<(3 * CH * CH + 255) / 256, 256, 0, stream>>>(w_qkv, wq_bf, 3 * CH * CH);
  cvt_bf16_kernel<<<(CH * CH + 255) / 256, 256, 0, stream>>>(w_out, wo_bf, CH * CH);

  groupnorm_kernel<<<BATCH * GROUPS, 256, 0, stream>>>(x, gamma, beta, xn_t);

  qkv_gemm_kernel<<<dim3(12, 8, BATCH), 128, 0, stream>>>(
      (const unsigned int*)wq_bf, (const unsigned int*)xn_t, b_qkv, qb, kb, vb, scale);

  attn_kernel<<<dim3(HW / 64, BATCH * HEADS), 128, 0, stream>>>(
      (const unsigned int*)qb, (const unsigned int*)kb, (const unsigned int*)vb, ao);

  out_gemm_kernel<<<dim3(4, 8, BATCH), 128, 0, stream>>>(
      (const unsigned int*)wo_bf, (const unsigned int*)ao, b_out, x, out);
}
